// HierarchicalSyntaxSemanticGNN_84550726189261
// MI455X (gfx1250) — compile-verified, matrix operand order checked by reading, not run on hardware
//
#include <hip/hip_runtime.h>
#include <hip/hip_bf16.h>

typedef __attribute__((ext_vector_type(16))) _Float16 v16h;
typedef __attribute__((ext_vector_type(8)))  float    v8f;
typedef __attribute__((ext_vector_type(4)))  unsigned tdm_v4u;
typedef __attribute__((ext_vector_type(8)))  int      tdm_v8i;
typedef __attribute__((ext_vector_type(4)))  int      tdm_v4i;

#define NN      4096
#define EE      32768
#define DIN     128
#define HDIM    256
#define OUTC    64
#define NHEADS  8
#define HEAD_D  32
#define EDGED   64
#define NG      16
#define NL      4
#define ECHUNK  4096

#define BM 64
#define BN 64
#define BK 32

#if __has_builtin(__builtin_amdgcn_tensor_load_to_lds) && \
    __has_builtin(__builtin_amdgcn_s_wait_tensorcnt)
#define HAVE_TDM 1
#else
#define HAVE_TDM 0
#endif

#if HAVE_TDM
// Build a 2D-tile D# per CDNA5 ISA ch.8 and issue TENSOR_LOAD_TO_LDS.
// tile_d0 = contiguous-dim length (elems), tile_d1 = rows, stride in elems.
// 6-arg builtin form (clang-23 / therock-10.0): (v4u, v8i, v4i, v4i, v8i, cpol)
__device__ __forceinline__ void tdm_load_2d(unsigned lds_addr, const float* gptr,
                                            int tile_d0, int tile_d1, int stride)
{
  unsigned long long ga = (unsigned long long)(const void*)gptr;
  tdm_v4u g0;
  g0[0] = 1u;                                   // count=1, user mode
  g0[1] = lds_addr;                             // LDS byte address
  g0[2] = (unsigned)ga;                         // global_addr[31:0]
  g0[3] = (unsigned)((ga >> 32) & 0x1FFFFFFu) | (2u << 30);  // addr[56:32] | type=2
  tdm_v8i g1;
  g1[0] = (int)(2u << 16);                      // data_size = 4B; no multicast/pad
  g1[1] = (int)(((unsigned)tile_d0 & 0xFFFFu) << 16);          // tensor_dim0 lo
  g1[2] = (int)((((unsigned)tile_d0 >> 16) & 0xFFFFu) |
                (((unsigned)tile_d1 & 0xFFFFu) << 16));        // d0 hi | d1 lo
  g1[3] = (int)((((unsigned)tile_d1 >> 16) & 0xFFFFu) |
                (((unsigned)tile_d0 & 0xFFFFu) << 16));        // d1 hi | tile_dim0
  g1[4] = (int)((unsigned)tile_d1 & 0xFFFFu);                  // tile_dim1 | tile_dim2=0
  g1[5] = stride;                               // tensor_dim0_stride[31:0]
  g1[6] = 0;                                    // stride hi | dim1_stride lo
  g1[7] = 0;
  tdm_v4i gz4 = {0, 0, 0, 0};
  tdm_v8i gz8 = {0, 0, 0, 0, 0, 0, 0, 0};
  __builtin_amdgcn_tensor_load_to_lds(g0, g1, gz4, gz4, gz8, 0);
}
#endif

// ---------------------------------------------------------------------------
// Tiled GEMM: C[M x Nc] = act(A[M x K] @ B[K x Nc] + bias), fp32 in/out,
// f16 WMMA compute with fp32 accumulate. Block = 256 threads (8 waves),
// 64x64 output tile, BK=32 k-strip staged in LDS (TDM when available).
// Requires M%64==0, K%32==0, Nc%64==0.
// ---------------------------------------------------------------------------
__global__ __launch_bounds__(256) void gemm_wmma(
    const float* __restrict__ A, const float* __restrict__ B,
    const float* __restrict__ bias, float* __restrict__ C,
    int M, int K, int Nc, int relu)
{
  __shared__ float As[BM][BK];   // [m][k]
  __shared__ float Bs[BK][BN];   // [k][n]

  const int tid  = threadIdx.x;
  const int wid  = tid >> 5;
  const int lane = tid & 31;
  const int hl   = lane >> 4;
  const int l15  = lane & 15;
  const int m_block = blockIdx.y * BM;
  const int n_block = blockIdx.x * BN;
  const int n_sub   = (wid & 3) * 16;       // wave's 16-col subtile
  const int m_sub0  = (wid >> 2) * 32;      // wave's two 16-row subtiles

  v8f acc0 = {0.f,0.f,0.f,0.f,0.f,0.f,0.f,0.f};
  v8f acc1 = acc0;

  for (int k0 = 0; k0 < K; k0 += BK) {
#if HAVE_TDM
    if (wid == 0) {
      tdm_load_2d((unsigned)(uintptr_t)&As[0][0],
                  A + (size_t)m_block * K + k0, BK, BM, K);
      tdm_load_2d((unsigned)(uintptr_t)&Bs[0][0],
                  B + (size_t)k0 * Nc + n_block, BN, BK, Nc);
      __builtin_amdgcn_s_wait_tensorcnt(0);
    }
    if (wid == 1 && k0 + BK < K) {
      __builtin_prefetch(A + (size_t)m_block * K + k0 + BK, 0, 1);
      __builtin_prefetch(B + (size_t)(k0 + BK) * Nc + n_block, 0, 1);
    }
#else
    for (int idx = tid; idx < (BM * BK) / 4; idx += 256) {
      const int r = idx / (BK / 4), cq = idx % (BK / 4);
      *(float4*)&As[r][cq * 4] =
          *(const float4*)(A + (size_t)(m_block + r) * K + k0 + cq * 4);
    }
    for (int idx = tid; idx < (BK * BN) / 4; idx += 256) {
      const int r = idx / (BN / 4), cq = idx % (BN / 4);
      *(float4*)&Bs[r][cq * 4] =
          *(const float4*)(B + (size_t)(k0 + r) * Nc + n_block + cq * 4);
    }
#endif
    __syncthreads();

    // B fragment (shared by both row subtiles): column n_sub+l15, strided k.
    v16h bfr;
#pragma unroll
    for (int j = 0; j < 16; ++j) {
      const int kk = 8 * hl + (j & 7) + ((j >> 3) << 4);
      bfr[j] = (_Float16)Bs[kk][n_sub + l15];
    }
    // A fragments: contiguous 8-float runs -> ds_load_b128 pairs.
#pragma unroll
    for (int mm = 0; mm < 2; ++mm) {
      const int row = m_sub0 + mm * 16 + l15;
      const float4 f0 = *(const float4*)&As[row][8 * hl];
      const float4 f1 = *(const float4*)&As[row][8 * hl + 4];
      const float4 f2 = *(const float4*)&As[row][16 + 8 * hl];
      const float4 f3 = *(const float4*)&As[row][16 + 8 * hl + 4];
      v16h afr;
      afr[0] = (_Float16)f0.x; afr[1] = (_Float16)f0.y;
      afr[2] = (_Float16)f0.z; afr[3] = (_Float16)f0.w;
      afr[4] = (_Float16)f1.x; afr[5] = (_Float16)f1.y;
      afr[6] = (_Float16)f1.z; afr[7] = (_Float16)f1.w;
      afr[8]  = (_Float16)f2.x; afr[9]  = (_Float16)f2.y;
      afr[10] = (_Float16)f2.z; afr[11] = (_Float16)f2.w;
      afr[12] = (_Float16)f3.x; afr[13] = (_Float16)f3.y;
      afr[14] = (_Float16)f3.z; afr[15] = (_Float16)f3.w;
      if (mm == 0)
        acc0 = __builtin_amdgcn_wmma_f32_16x16x32_f16(false, afr, false, bfr,
                                                      (short)0, acc0, false, false);
      else
        acc1 = __builtin_amdgcn_wmma_f32_16x16x32_f16(false, afr, false, bfr,
                                                      (short)0, acc1, false, false);
    }
    __syncthreads();
  }

  const int n = n_block + n_sub + l15;
  const float bv = bias ? bias[n] : 0.0f;
#pragma unroll
  for (int r = 0; r < 8; ++r) {
    const int m0 = m_block + m_sub0 + r + 8 * hl;
    float v0 = acc0[r] + bv;
    float v1 = acc1[r] + bv;
    if (relu) { v0 = fmaxf(v0, 0.0f); v1 = fmaxf(v1, 0.0f); }
    C[(size_t)m0 * Nc + n] = v0;
    C[(size_t)(m0 + 16) * Nc + n] = v1;
  }
}

// ---------------------------------------------------------------------------
// MHA pass 1: per-(head,row) online softmax stats (max, sumexp) via WMMA
// score tiles. Grid: (N/16, HEADS), block 32 (one wave).
// ---------------------------------------------------------------------------
__global__ __launch_bounds__(32) void mha_stats(
    const float* __restrict__ q, const float* __restrict__ k,
    float* __restrict__ rowmax, float* __restrict__ rowsum)
{
  const int h  = blockIdx.y;
  const int n0 = blockIdx.x * 16;
  const int lane = threadIdx.x, hl = lane >> 4, l15 = lane & 15;
  const float scale = 0.17677669529663689f;   // 1/sqrt(32)

  v16h a;
  const float* Qrow = q + (size_t)(n0 + l15) * HDIM + h * HEAD_D + 8 * hl;
#pragma unroll
  for (int j = 0; j < 16; ++j) a[j] = (_Float16)Qrow[(j & 7) + ((j >> 3) << 4)];

  float rmax[8], rsum[8];
#pragma unroll
  for (int r = 0; r < 8; ++r) { rmax[r] = -1e30f; rsum[r] = 0.0f; }

  for (int m0 = 0; m0 < NN; m0 += 16) {
    v16h b;
    const float* Krow = k + (size_t)(m0 + l15) * HDIM + h * HEAD_D + 8 * hl;
#pragma unroll
    for (int j = 0; j < 16; ++j) b[j] = (_Float16)Krow[(j & 7) + ((j >> 3) << 4)];
    v8f s = {0.f,0.f,0.f,0.f,0.f,0.f,0.f,0.f};
    s = __builtin_amdgcn_wmma_f32_16x16x32_f16(false, a, false, b,
                                               (short)0, s, false, false);
#pragma unroll
    for (int r = 0; r < 8; ++r) {
      const float v = s[r] * scale;
      float tm = v;
      for (int w = 1; w < 16; w <<= 1) tm = fmaxf(tm, __shfl_xor(tm, w, 32));
      const float nm = fmaxf(rmax[r], tm);
      float e = __expf(v - nm);
      for (int w = 1; w < 16; w <<= 1) e += __shfl_xor(e, w, 32);
      rsum[r] = rsum[r] * __expf(rmax[r] - nm) + e;
      rmax[r] = nm;
    }
  }
  if (l15 == 0) {
#pragma unroll
    for (int r = 0; r < 8; ++r) {
      const int idx = h * NN + n0 + r + 8 * hl;
      rowmax[idx] = rmax[r];
      rowsum[idx] = rsum[r];
    }
  }
}

// ---------------------------------------------------------------------------
// MHA pass 2a: joint = attn @ V, streamed over key chunks of 32.
// Grid: (N/16, HEADS), block 32.
// ---------------------------------------------------------------------------
__global__ __launch_bounds__(32) void mha_joint(
    const float* __restrict__ q, const float* __restrict__ k,
    const float* __restrict__ v, const float* __restrict__ rowmax,
    const float* __restrict__ rowsum, float* __restrict__ joint)
{
  __shared__ float tile[16 * 32];
  const int h  = blockIdx.y;
  const int n0 = blockIdx.x * 16;
  const int lane = threadIdx.x, hl = lane >> 4, l15 = lane & 15;
  const float scale = 0.17677669529663689f;

  v16h a;
  const float* Qrow = q + (size_t)(n0 + l15) * HDIM + h * HEAD_D + 8 * hl;
#pragma unroll
  for (int j = 0; j < 16; ++j) a[j] = (_Float16)Qrow[(j & 7) + ((j >> 3) << 4)];

  float rmax[8], rsi[8];
#pragma unroll
  for (int r = 0; r < 8; ++r) {
    const int idx = h * NN + n0 + r + 8 * hl;
    rmax[r] = rowmax[idx];
    rsi[r]  = 1.0f / rowsum[idx];
  }

  v8f acc0 = {0.f,0.f,0.f,0.f,0.f,0.f,0.f,0.f};
  v8f acc1 = acc0;

  for (int m0 = 0; m0 < NN; m0 += 32) {
#pragma unroll
    for (int t = 0; t < 2; ++t) {
      v16h b;
      const float* Krow = k + (size_t)(m0 + 16 * t + l15) * HDIM + h * HEAD_D + 8 * hl;
#pragma unroll
      for (int j = 0; j < 16; ++j) b[j] = (_Float16)Krow[(j & 7) + ((j >> 3) << 4)];
      v8f s = {0.f,0.f,0.f,0.f,0.f,0.f,0.f,0.f};
      s = __builtin_amdgcn_wmma_f32_16x16x32_f16(false, a, false, b,
                                                 (short)0, s, false, false);
#pragma unroll
      for (int r = 0; r < 8; ++r) {
        const float p = __expf(s[r] * scale - rmax[r]) * rsi[r];
        tile[(r + 8 * hl) * 32 + 16 * t + l15] = p;
      }
    }
    __syncthreads();
    v16h a2;
#pragma unroll
    for (int j = 0; j < 16; ++j)
      a2[j] = (_Float16)tile[l15 * 32 + 8 * hl + (j & 7) + ((j >> 3) << 4)];
#pragma unroll
    for (int t2 = 0; t2 < 2; ++t2) {
      v16h b2;
#pragma unroll
      for (int j = 0; j < 16; ++j) {
        const int kk = 8 * hl + (j & 7) + ((j >> 3) << 4);
        b2[j] = (_Float16)v[(size_t)(m0 + kk) * HDIM + h * HEAD_D + 16 * t2 + l15];
      }
      if (t2 == 0)
        acc0 = __builtin_amdgcn_wmma_f32_16x16x32_f16(false, a2, false, b2,
                                                      (short)0, acc0, false, false);
      else
        acc1 = __builtin_amdgcn_wmma_f32_16x16x32_f16(false, a2, false, b2,
                                                      (short)0, acc1, false, false);
    }
    __syncthreads();
  }
#pragma unroll
  for (int r = 0; r < 8; ++r) {
    const int row = n0 + r + 8 * hl;
    joint[(size_t)row * HDIM + h * HEAD_D + l15]      = acc0[r];
    joint[(size_t)row * HDIM + h * HEAD_D + 16 + l15] = acc1[r];
  }
}

// ---------------------------------------------------------------------------
// MHA pass 2b: attn_w = mean over heads of softmax(scores). One 16x16 output
// tile per wave; 8 score WMMAs (one per head). Grid: (N/16, N/16).
// ---------------------------------------------------------------------------
__global__ __launch_bounds__(32) void mha_attnw(
    const float* __restrict__ q, const float* __restrict__ k,
    const float* __restrict__ rowmax, const float* __restrict__ rowsum,
    float* __restrict__ attnw)
{
  const int m0 = blockIdx.x * 16;   // key (column) tile
  const int n0 = blockIdx.y * 16;   // query (row) tile
  const int lane = threadIdx.x, hl = lane >> 4, l15 = lane & 15;
  const float scale = 0.17677669529663689f;

  float mean[8] = {0.f,0.f,0.f,0.f,0.f,0.f,0.f,0.f};
  for (int h = 0; h < NHEADS; ++h) {
    v16h a, b;
    const float* Qrow = q + (size_t)(n0 + l15) * HDIM + h * HEAD_D + 8 * hl;
    const float* Krow = k + (size_t)(m0 + l15) * HDIM + h * HEAD_D + 8 * hl;
#pragma unroll
    for (int j = 0; j < 16; ++j) {
      const int kk = (j & 7) + ((j >> 3) << 4);
      a[j] = (_Float16)Qrow[kk];
      b[j] = (_Float16)Krow[kk];
    }
    v8f s = {0.f,0.f,0.f,0.f,0.f,0.f,0.f,0.f};
    s = __builtin_amdgcn_wmma_f32_16x16x32_f16(false, a, false, b,
                                               (short)0, s, false, false);
#pragma unroll
    for (int r = 0; r < 8; ++r) {
      const int idx = h * NN + n0 + r + 8 * hl;
      mean[r] += __expf(s[r] * scale - rowmax[idx]) / rowsum[idx];
    }
  }
#pragma unroll
  for (int r = 0; r < 8; ++r)
    attnw[(size_t)(n0 + r + 8 * hl) * NN + m0 + l15] = mean[r] * 0.125f;
}

// ---------------------------------------------------------------------------
// Edge kernels
// ---------------------------------------------------------------------------
__device__ __forceinline__ unsigned float_key(float f) {
  unsigned b = __float_as_uint(f);
  return (b & 0x80000000u) ? ~b : (b | 0x80000000u);
}
__device__ __forceinline__ float key_float(unsigned key) {
  unsigned b = (key & 0x80000000u) ? (key ^ 0x80000000u) : ~key;
  return __uint_as_float(b);
}

__global__ __launch_bounds__(256) void edge_logits(
    const int* __restrict__ src, const int* __restrict__ dst,
    const float* __restrict__ qi, const float* __restrict__ ki,
    const float* __restrict__ eproj, int chunk_base, int chunk_E,
    float* __restrict__ logits, unsigned* __restrict__ maxkey)
{
  const int wid  = (blockIdx.x * blockDim.x + threadIdx.x) >> 5;
  const int lane = threadIdx.x & 31;
  if (wid >= chunk_E * NHEADS) return;
  const int le = wid >> 3, hh = wid & 7;
  const int e  = chunk_base + le;
  const int sN = src[e], dN = dst[e];
  const float* qp = qi + (size_t)dN * (NHEADS * HDIM) + hh * HDIM;
  const float* kp = ki + (size_t)sN * (NHEADS * HDIM) + hh * HDIM;
  const float* ep = eproj + (size_t)le * (NHEADS * HDIM) + hh * HDIM;
  float acc = 0.0f;
  for (int t = lane; t < HDIM; t += 32) acc += qp[t] * (kp[t] + ep[t]);
  for (int w = 1; w < 32; w <<= 1) acc += __shfl_xor(acc, w, 32);
  if (lane == 0) {
    const float lg = acc * 0.0625f;            // 1/sqrt(256)
    logits[(size_t)e * NHEADS + hh] = lg;
    atomicMax(&maxkey[dN * NHEADS + hh], float_key(lg));
  }
}

__global__ __launch_bounds__(256) void edge_softmax_p(
    const int* __restrict__ dst, const float* __restrict__ logits,
    const unsigned* __restrict__ maxkey, float* __restrict__ p,
    float* __restrict__ denom)
{
  const int i = blockIdx.x * blockDim.x + threadIdx.x;
  if (i >= EE * NHEADS) return;
  const int e = i >> 3, hh = i & 7;
  const int dN = dst[e];
  const float m = key_float(maxkey[dN * NHEADS + hh]);
  const float pe = __expf(logits[i] - m);
  p[i] = pe;
  atomicAdd(&denom[dN * NHEADS + hh], pe);
}

__global__ __launch_bounds__(256) void edge_agg(
    const int* __restrict__ src, const int* __restrict__ dst,
    const float* __restrict__ vi, const float* __restrict__ eproj,
    const float* __restrict__ p, const float* __restrict__ denom,
    float* __restrict__ agg, int chunk_base)
{
  const int le = blockIdx.x >> 3, hh = blockIdx.x & 7;
  const int e  = chunk_base + le;
  const int d  = threadIdx.x;                 // 0..255
  const int sN = src[e], dN = dst[e];
  const float alpha = p[(size_t)e * NHEADS + hh] / denom[dN * NHEADS + hh];
  const float val = (vi[(size_t)sN * (NHEADS * HDIM) + hh * HDIM + d] +
                     eproj[(size_t)le * (NHEADS * HDIM) + hh * HDIM + d]) * alpha;
  atomicAdd(&agg[(size_t)dN * (NHEADS * HDIM) + hh * HDIM + d], val);
}

__global__ __launch_bounds__(256) void combine_heads(
    const float* __restrict__ agg, const float* __restrict__ skip,
    float* __restrict__ out)
{
  const int i = blockIdx.x * blockDim.x + threadIdx.x;
  if (i >= NN * HDIM) return;
  const int n = i >> 8, d = i & 255;
  float s = 0.0f;
#pragma unroll
  for (int hh = 0; hh < NHEADS; ++hh)
    s += agg[(size_t)n * (NHEADS * HDIM) + hh * HDIM + d];
  out[i] = s * 0.125f + skip[i];
}

__global__ __launch_bounds__(256) void concat2(
    const float* __restrict__ a, const float* __restrict__ b,
    float* __restrict__ c)
{
  const int i = blockIdx.x * blockDim.x + threadIdx.x;
  if (i >= NN * 2 * HDIM) return;
  const int n = i / (2 * HDIM), d = i % (2 * HDIM);
  c[i] = (d < HDIM) ? a[(size_t)n * HDIM + d] : b[(size_t)n * HDIM + d - HDIM];
}

__global__ __launch_bounds__(256) void copy_f32(
    const float* __restrict__ a, float* __restrict__ b, int n)
{
  const int i = blockIdx.x * blockDim.x + threadIdx.x;
  if (i < n) b[i] = a[i];
}

__global__ __launch_bounds__(256) void pool_sum(
    const float* __restrict__ hmat, const int* __restrict__ batch,
    float* __restrict__ gsum, float* __restrict__ counts)
{
  const int i = blockIdx.x * blockDim.x + threadIdx.x;
  if (i >= NN * HDIM) return;
  const int n = i >> 8, d = i & 255;
  const int g = batch[n];
  atomicAdd(&gsum[(size_t)g * HDIM + d], hmat[i]);
  if (d == 0) atomicAdd(&counts[g], 1.0f);
}

__global__ __launch_bounds__(256) void pool_div(
    float* __restrict__ gsum, const float* __restrict__ counts)
{
  const int i = blockIdx.x * blockDim.x + threadIdx.x;
  if (i >= NG * HDIM) return;
  gsum[i] /= fmaxf(counts[i >> 8], 1.0f);
}

__global__ __launch_bounds__(256) void vul_head(
    const float* __restrict__ hh, const float* __restrict__ W,
    const float* __restrict__ b, float* __restrict__ out)
{
  const int n = blockIdx.x * blockDim.x + threadIdx.x;
  if (n >= NN) return;
  float acc = b[0];
#pragma unroll 8
  for (int j = 0; j < 128; ++j) acc += hh[(size_t)n * 128 + j] * W[j];
  out[n] = 1.0f / (1.0f + __expf(-acc));
}

// ---------------------------------------------------------------------------
// Host orchestration
// ---------------------------------------------------------------------------
static inline void launch_gemm(const float* A, const float* B, const float* bias,
                               float* C, int M, int K, int Nc, int relu,
                               hipStream_t s)
{
  dim3 grid(Nc / BN, M / BM);
  gemm_wmma<<<grid, 256, 0, s>>>(A, B, bias, C, M, K, Nc, relu);
}

extern "C" void kernel_launch(void* const* d_in, const int* in_sizes, int n_in,
                              void* d_out, int out_size, void* d_ws, size_t ws_size,
                              hipStream_t stream) {
  const float* x         = (const float*)d_in[0];
  const int*   edge_idx  = (const int*)d_in[1];
  const float* edge_attr = (const float*)d_in[2];
  const int*   batch     = (const int*)d_in[3];
  const float* syn_W1 = (const float*)d_in[4];  const float* syn_b1 = (const float*)d_in[5];
  const float* syn_W2 = (const float*)d_in[6];  const float* syn_b2 = (const float*)d_in[7];
  const float* sem_W1 = (const float*)d_in[8];  const float* sem_b1 = (const float*)d_in[9];
  const float* sem_W2 = (const float*)d_in[10]; const float* sem_b2 = (const float*)d_in[11];
  const float* mha_Wq = (const float*)d_in[12]; const float* mha_bq = (const float*)d_in[13];
  const float* mha_Wk = (const float*)d_in[14]; const float* mha_bk = (const float*)d_in[15];
  const float* mha_Wv = (const float*)d_in[16]; const float* mha_bv = (const float*)d_in[17];
  const float* mha_Wo = (const float*)d_in[18]; const float* mha_bo = (const float*)d_in[19];
  const float* conv_Wq = (const float*)d_in[20]; const float* conv_bq = (const float*)d_in[21];
  const float* conv_Wk = (const float*)d_in[22]; const float* conv_bk = (const float*)d_in[23];
  const float* conv_Wv = (const float*)d_in[24]; const float* conv_bv = (const float*)d_in[25];
  const float* conv_We = (const float*)d_in[26]; const float* conv_be = (const float*)d_in[27];
  const float* conv_Ws = (const float*)d_in[28]; const float* conv_bs = (const float*)d_in[29];
  const float* fus_W   = (const float*)d_in[30]; const float* fus_b   = (const float*)d_in[31];
  const float* cls_W1  = (const float*)d_in[32]; const float* cls_b1  = (const float*)d_in[33];
  const float* cls_W2  = (const float*)d_in[34]; const float* cls_b2  = (const float*)d_in[35];
  const float* vul_W1  = (const float*)d_in[36]; const float* vul_b1  = (const float*)d_in[37];
  const float* vul_W2  = (const float*)d_in[38]; const float* vul_b2  = (const float*)d_in[39];

  const int* e_src = edge_idx;
  const int* e_dst = edge_idx + EE;

  float* out_h    = (float*)d_out;                  // 4096*256
  float* out_ge   = out_h  + (size_t)NN * HDIM;     // 16*256
  float* out_cls  = out_ge + (size_t)NG * HDIM;     // 4096*64
  float* out_vul  = out_cls + (size_t)NN * OUTC;    // 4096
  float* out_attn = out_vul + NN;                   // 4096*4096

  char* wp = (char*)d_ws;
  auto alloc = [&](size_t nfloats) -> float* {
    float* p = (float*)wp;
    wp += ((nfloats * sizeof(float) + 255) / 256) * 256;
    return p;
  };
  float* t1     = alloc((size_t)NN * HDIM);
  float* syn    = alloc((size_t)NN * HDIM);
  float* sem    = alloc((size_t)NN * HDIM);
  float* qb     = alloc((size_t)NN * HDIM);
  float* kb     = alloc((size_t)NN * HDIM);
  float* vb     = alloc((size_t)NN * HDIM);
  float* rowmax = alloc((size_t)NHEADS * NN);
  float* rowsum = alloc((size_t)NHEADS * NN);
  float* jointb = alloc((size_t)NN * HDIM);
  float* hbuf0  = alloc((size_t)NN * HDIM);
  float* hbuf1  = alloc((size_t)NN * HDIM);
  float* obuf   = alloc((size_t)NN * HDIM);
  float* catbuf = alloc((size_t)NN * 2 * HDIM);
  float* qi     = alloc((size_t)NN * NHEADS * HDIM);
  float* ki     = alloc((size_t)NN * NHEADS * HDIM);
  float* vi     = alloc((size_t)NN * NHEADS * HDIM);
  float* eproj  = alloc((size_t)ECHUNK * NHEADS * HDIM);
  float* logits = alloc((size_t)EE * NHEADS);
  float* pvals  = alloc((size_t)EE * NHEADS);
  unsigned* maxkey = (unsigned*)alloc((size_t)NN * NHEADS);
  float* denom  = alloc((size_t)NN * NHEADS);
  float* agg    = alloc((size_t)NN * NHEADS * HDIM);
  float* skip   = alloc((size_t)NN * HDIM);
  float* counts = alloc(NG);
  float* clsh   = alloc((size_t)NN * 128);
  float* vulh   = alloc((size_t)NN * 128);
  (void)ws_size; (void)n_in; (void)in_sizes; (void)out_size;

  // ---- Encoders ----
  launch_gemm(x,  syn_W1, syn_b1, t1,  NN, DIN,  HDIM, 1, stream);
  launch_gemm(t1, syn_W2, syn_b2, syn, NN, HDIM, HDIM, 0, stream);
  launch_gemm(x,  sem_W1, sem_b1, t1,  NN, DIN,  HDIM, 1, stream);
  launch_gemm(t1, sem_W2, sem_b2, sem, NN, HDIM, HDIM, 0, stream);

  // ---- Joint MHA ----
  launch_gemm(syn, mha_Wq, mha_bq, qb, NN, HDIM, HDIM, 0, stream);
  launch_gemm(sem, mha_Wk, mha_bk, kb, NN, HDIM, HDIM, 0, stream);
  launch_gemm(sem, mha_Wv, mha_bv, vb, NN, HDIM, HDIM, 0, stream);
  mha_stats<<<dim3(NN / 16, NHEADS), 32, 0, stream>>>(qb, kb, rowmax, rowsum);
  mha_joint<<<dim3(NN / 16, NHEADS), 32, 0, stream>>>(qb, kb, vb, rowmax, rowsum, jointb);
  mha_attnw<<<dim3(NN / 16, NN / 16), 32, 0, stream>>>(qb, kb, rowmax, rowsum, out_attn);
  launch_gemm(jointb, mha_Wo, mha_bo, hbuf0, NN, HDIM, HDIM, 0, stream);

  // ---- TransformerConv layers ----
  float* hcur = hbuf0;
  float* hnext = hbuf1;
  const int NWIDE = NHEADS * HDIM;   // 2048
  for (int i = 0; i < NL; ++i) {
    const float* Wq = conv_Wq + (size_t)i * HDIM * NWIDE;
    const float* Wk = conv_Wk + (size_t)i * HDIM * NWIDE;
    const float* Wv = conv_Wv + (size_t)i * HDIM * NWIDE;
    const float* We = conv_We + (size_t)i * EDGED * NWIDE;
    const float* Ws = conv_Ws + (size_t)i * HDIM * HDIM;
    const float* bq = conv_bq + (size_t)i * NWIDE;
    const float* bk = conv_bk + (size_t)i * NWIDE;
    const float* bv = conv_bv + (size_t)i * NWIDE;
    const float* be = conv_be + (size_t)i * NWIDE;
    const float* bs = conv_bs + (size_t)i * HDIM;

    launch_gemm(hcur, Wq, bq, qi, NN, HDIM, NWIDE, 0, stream);
    launch_gemm(hcur, Wk, bk, ki, NN, HDIM, NWIDE, 0, stream);
    launch_gemm(hcur, Wv, bv, vi, NN, HDIM, NWIDE, 0, stream);
    launch_gemm(hcur, Ws, bs, skip, NN, HDIM, HDIM, 0, stream);

    (void)hipMemsetAsync(maxkey, 0, (size_t)NN * NHEADS * sizeof(unsigned), stream);
    (void)hipMemsetAsync(denom, 0, (size_t)NN * NHEADS * sizeof(float), stream);
    (void)hipMemsetAsync(agg, 0, (size_t)NN * NHEADS * HDIM * sizeof(float), stream);

    for (int c = 0; c < EE; c += ECHUNK) {
      launch_gemm(edge_attr + (size_t)c * EDGED, We, be, eproj,
                  ECHUNK, EDGED, NWIDE, 0, stream);
      const int nwaves = ECHUNK * NHEADS;
      edge_logits<<<(nwaves * 32 + 255) / 256, 256, 0, stream>>>(
          e_src, e_dst, qi, ki, eproj, c, ECHUNK, logits, maxkey);
    }
    edge_softmax_p<<<(EE * NHEADS + 255) / 256, 256, 0, stream>>>(
        e_dst, logits, maxkey, pvals, denom);
    for (int c = 0; c < EE; c += ECHUNK) {
      launch_gemm(edge_attr + (size_t)c * EDGED, We, be, eproj,
                  ECHUNK, EDGED, NWIDE, 0, stream);
      edge_agg<<<ECHUNK * NHEADS, 256, 0, stream>>>(
          e_src, e_dst, vi, eproj, pvals, denom, agg, c);
    }
    combine_heads<<<(NN * HDIM + 255) / 256, 256, 0, stream>>>(agg, skip, obuf);

    if (i == 0) {
      copy_f32<<<(NN * HDIM + 255) / 256, 256, 0, stream>>>(obuf, hnext, NN * HDIM);
    } else {
      concat2<<<(NN * 2 * HDIM + 255) / 256, 256, 0, stream>>>(obuf, hcur, catbuf);
      launch_gemm(catbuf, fus_W + (size_t)i * 2 * HDIM * HDIM,
                  fus_b + (size_t)i * HDIM, hnext, NN, 2 * HDIM, HDIM, 1, stream);
    }
    float* tmp = hcur; hcur = hnext; hnext = tmp;
  }

  // ---- Outputs ----
  copy_f32<<<(NN * HDIM + 255) / 256, 256, 0, stream>>>(hcur, out_h, NN * HDIM);

  (void)hipMemsetAsync(out_ge, 0, (size_t)NG * HDIM * sizeof(float), stream);
  (void)hipMemsetAsync(counts, 0, NG * sizeof(float), stream);
  pool_sum<<<(NN * HDIM + 255) / 256, 256, 0, stream>>>(hcur, batch, out_ge, counts);
  pool_div<<<(NG * HDIM + 255) / 256, 256, 0, stream>>>(out_ge, counts);

  launch_gemm(hcur, cls_W1, cls_b1, clsh, NN, HDIM, 128, 1, stream);
  launch_gemm(clsh, cls_W2, cls_b2, out_cls, NN, 128, OUTC, 0, stream);

  launch_gemm(hcur, vul_W1, vul_b1, vulh, NN, HDIM, 128, 1, stream);
  vul_head<<<(NN + 255) / 256, 256, 0, stream>>>(vulh, vul_W2, vul_b2, out_vul);
}
